// Attention_78898549227928
// MI455X (gfx1250) — matrix operand order repaired and checked by reading, not compiled, not run on hardware
//
#include <hip/hip_runtime.h>

typedef __attribute__((ext_vector_type(16))) _Float16 v16h;
typedef __attribute__((ext_vector_type(8)))  float    v8f;

union AF { v16h h; uint4 q[2]; unsigned u[8]; };

#if __has_builtin(__builtin_amdgcn_cvt_pkrtz)
__device__ __forceinline__ unsigned pk2(float a, float b) {
  auto r = __builtin_amdgcn_cvt_pkrtz(a, b);          // v_cvt_pk_rtz_f16_f32
  return __builtin_bit_cast(unsigned, r);
}
#else
__device__ __forceinline__ unsigned pk2(float a, float b) {
  unsigned short ua = __builtin_bit_cast(unsigned short, (_Float16)a);
  unsigned short ub = __builtin_bit_cast(unsigned short, (_Float16)b);
  return (unsigned)ua | ((unsigned)ub << 16);
}
#endif

// CDNA5 async global->LDS path (ASYNCcnt), with safe fallback.
// Param types per hipcc diagnostic: (int4 AS1*, int4 AS3*, imm, imm)
#if __has_builtin(__builtin_amdgcn_global_load_async_to_lds_b128) && \
    __has_builtin(__builtin_amdgcn_s_wait_asynccnt)
#define USE_ASYNC 1
typedef __attribute__((__vector_size__(4 * sizeof(int)))) int vs4i;
__device__ __forceinline__ void async_cp16(void* lds, const void* g) {
  __builtin_amdgcn_global_load_async_to_lds_b128(
      (__attribute__((address_space(1))) vs4i*)g,
      (__attribute__((address_space(3))) vs4i*)lds, 0, 0);
}
#else
#define USE_ASYNC 0
#endif

#define NB 4
#define CH 128
#define CO 64
#define NN 4096
#define NITER (NN / 32)

// padded LDS row strides (f16 units) chosen for conflict-free b128 row reads
#define XS 136   // 64x(128+8)  : x^T and weight tiles
#define KS 72    // 32x(64+8)   : K tile
#define VSS 40   // 64x(32+8)   : V tile

// ---------------------------------------------------------------------------
// Projection compute+store, P resolved at compile time.
// Dt[o,n] = sum_c w[o,c] * x[c,n] + bias[o].
// P==0 -> Qh[b][n][o], P==1 -> Kh[b][n][o], P==2 -> Vt[b][o][n]
// ---------------------------------------------------------------------------
template<int P>
__device__ __forceinline__ void proj_cs(
    const _Float16* w_s, const float* bias_s, const AF* Bf,
    _Float16* __restrict__ dst, int b, int n, int l16, bool hi)
{
#pragma unroll
  for (int ot = 0; ot < 4; ++ot) {
    v8f acc = {};
#pragma unroll
    for (int cc = 0; cc < 4; ++cc) {
      AF Af;
      int row = ot * 16 + l16;
      int eb  = cc * 32 + (hi ? 8 : 0);
      Af.q[0] = *(const uint4*)&w_s[row * XS + eb];
      Af.q[1] = *(const uint4*)&w_s[row * XS + eb + 16];
      acc = __builtin_amdgcn_wmma_f32_16x16x32_f16(false, Af.h, false, Bf[cc].h,
                                                   (short)0, acc, false, false);
    }
#pragma unroll
    for (int j = 0; j < 8; ++j) {
      int o = ot * 16 + j + (hi ? 8 : 0);
      _Float16 hv = (_Float16)(acc[j] + bias_s[o]);
      if constexpr (P == 2) dst[((size_t)(b * CO + o)) * NN + n] = hv;
      else                  dst[((size_t)b * NN + n) * CO + o]   = hv;
    }
  }
}

__device__ __forceinline__ void build_Bf(AF* Bf, const _Float16* x_t,
                                         int nl, bool hi)
{
#pragma unroll
  for (int cc = 0; cc < 4; ++cc) {
    int cb = cc * 32 + (hi ? 8 : 0);
    const uint4* p = (const uint4*)&x_t[nl * XS + cb];
    Bf[cc].q[0] = p[0];
    Bf[cc].q[1] = p[2];
  }
}

__global__ __launch_bounds__(128) void proj_kernel(
    const float* __restrict__ range_x, const float* __restrict__ img,
    const float* __restrict__ wq, const float* __restrict__ bq,
    const float* __restrict__ wk, const float* __restrict__ bk,
    const float* __restrict__ wv, const float* __restrict__ bv,
    _Float16* __restrict__ Qh, _Float16* __restrict__ Kh, _Float16* __restrict__ Vt)
{
  __shared__ __align__(16) _Float16 w_s[CO * XS];
  __shared__ __align__(16) _Float16 x_t[64 * XS];   // [n_local][c] transposed
  __shared__ float bias_s[CO];

  const int tid  = threadIdx.x;
  const int b    = blockIdx.x >> 6;
  const int n0   = (blockIdx.x & 63) * 64;
  const int wave = tid >> 5;
  const int lane = tid & 31;
  const int l16  = lane & 15;
  const bool hi  = lane >= 16;
  const int nl   = wave * 16 + l16;
  const int n    = n0 + nl;

  AF Bf[4];

  // ---- pass 1: img -> K and V (x staged once, Bf reused) ----
  for (int i = tid; i < CH * 64; i += 128) {
    int c = i >> 6, nn = i & 63;
    x_t[nn * XS + c] = (_Float16)img[((size_t)(b * CH + c)) * NN + n0 + nn];
  }
  for (int i = tid; i < CO * CH; i += 128)
    w_s[(i >> 7) * XS + (i & 127)] = (_Float16)wk[i];
  if (tid < CO) bias_s[tid] = bk[tid];
  __syncthreads();

  build_Bf(Bf, x_t, nl, hi);
  proj_cs<1>(w_s, bias_s, Bf, Kh, b, n, l16, hi);
  __syncthreads();

  for (int i = tid; i < CO * CH; i += 128)
    w_s[(i >> 7) * XS + (i & 127)] = (_Float16)wv[i];
  if (tid < CO) bias_s[tid] = bv[tid];
  __syncthreads();

  proj_cs<2>(w_s, bias_s, Bf, Vt, b, n, l16, hi);
  __syncthreads();

  // ---- pass 2: range_x -> Q ----
  for (int i = tid; i < CH * 64; i += 128) {
    int c = i >> 6, nn = i & 63;
    x_t[nn * XS + c] = (_Float16)range_x[((size_t)(b * CH + c)) * NN + n0 + nn];
  }
  for (int i = tid; i < CO * CH; i += 128)
    w_s[(i >> 7) * XS + (i & 127)] = (_Float16)wq[i];
  if (tid < CO) bias_s[tid] = bq[tid];
  __syncthreads();

  build_Bf(Bf, x_t, nl, hi);
  proj_cs<0>(w_s, bias_s, Bf, Qh, b, n, l16, hi);
}

// ---------------------------------------------------------------------------
// Fused flash attention + wc conv + BN + ReLU + residual.
// Double-buffered K/V LDS tiles; async global->LDS when available.
// ---------------------------------------------------------------------------
__global__ __launch_bounds__(128) void flash_kernel(
    const _Float16* __restrict__ Qh, const _Float16* __restrict__ Kh,
    const _Float16* __restrict__ Vt, const float* __restrict__ img,
    const float* __restrict__ wc, const float* __restrict__ bc,
    const float* __restrict__ gamma, const float* __restrict__ beta,
    const float* __restrict__ mean, const float* __restrict__ var,
    float* __restrict__ out)
{
  __shared__ __align__(16) _Float16 wc_s[CH * KS];
  __shared__ __align__(16) _Float16 Ksh[2][32 * KS];
  __shared__ __align__(16) _Float16 Vsh[2][CO * VSS];
  __shared__ float scale_s[CH], shift_s[CH];

  const int tid  = threadIdx.x;
  const int b    = blockIdx.x >> 6;
  const int q0   = (blockIdx.x & 63) * 64;
  const int wave = tid >> 5;
  const int lane = tid & 31;
  const int l16  = lane & 15;
  const bool hi  = lane >= 16;
  const int qn0  = q0 + wave * 16;

  for (int i = tid; i < CH * CO; i += 128)
    wc_s[(i >> 6) * KS + (i & 63)] = (_Float16)wc[i];
  if (tid < CH) {
    float inv = gamma[tid] * rsqrtf(var[tid] + 1e-5f);
    scale_s[tid] = inv;
    shift_s[tid] = beta[tid] - mean[tid] * inv + bc[tid] * inv;
  }

  // per-thread cooperative-load coordinates
  const int krow = tid >> 2, kcb = (tid & 3) * 16;
  const int vrow = tid >> 1, vcb = (tid & 1) * 16;
  const _Float16* kg = Kh + ((size_t)b * NN + krow) * CO + kcb;       // + m0*CO
  const _Float16* vg = Vt + ((size_t)(b * CO + vrow)) * NN + vcb;     // + m0
  _Float16* kbufs[2] = { &Ksh[0][krow * KS + kcb], &Ksh[1][krow * KS + kcb] };
  _Float16* vbufs[2] = { &Vsh[0][vrow * VSS + vcb], &Vsh[1][vrow * VSS + vcb] };

  // Q as B-operand fragments (col = query = lane%16), two 32-wide d-chunks.
  AF Qb[2];
  {
    size_t base = ((size_t)b * NN + qn0 + l16) * CO + (hi ? 8 : 0);
    const uint4* g0 = (const uint4*)(Qh + base);
    Qb[0].q[0] = g0[0]; Qb[0].q[1] = g0[2];
    const uint4* g1 = (const uint4*)(Qh + base + 32);
    Qb[1].q[0] = g1[0]; Qb[1].q[1] = g1[2];
  }

  float m_i = -3.0e38f, l_i = 0.f;
  v8f accO[4];
#pragma unroll
  for (int ot = 0; ot < 4; ++ot)
#pragma unroll
    for (int j = 0; j < 8; ++j) accO[ot][j] = 0.f;

  // ---- prologue: chunk 0 into buffer 0 ----
#if USE_ASYNC
  async_cp16(kbufs[0],     kg);
  async_cp16(kbufs[0] + 8, kg + 8);
  async_cp16(vbufs[0],     vg);
  async_cp16(vbufs[0] + 8, vg + 8);
  __builtin_amdgcn_s_wait_asynccnt(0);
#else
  {
    const uint4* g = (const uint4*)kg;
    ((uint4*)kbufs[0])[0] = g[0]; ((uint4*)kbufs[0])[1] = g[1];
    const uint4* h = (const uint4*)vg;
    ((uint4*)vbufs[0])[0] = h[0]; ((uint4*)vbufs[0])[1] = h[1];
  }
#endif
  __syncthreads();

  for (int kb = 0; kb < NITER; ++kb) {
    const int cur = kb & 1;
    const bool more = (kb + 1) < NITER;
    const _Float16* Kc = Ksh[cur];
    const _Float16* Vc = Vsh[cur];

#if USE_ASYNC
    if (more) {   // prefetch next chunk into the idle buffer (ASYNCcnt path)
      const _Float16* kgn = kg + (size_t)(kb + 1) * 32 * CO;
      const _Float16* vgn = vg + (size_t)(kb + 1) * 32;
      async_cp16(kbufs[cur ^ 1],     kgn);
      async_cp16(kbufs[cur ^ 1] + 8, kgn + 8);
      async_cp16(vbufs[cur ^ 1],     vgn);
      async_cp16(vbufs[cur ^ 1] + 8, vgn + 8);
    }
#else
    uint4 k0, k1, v0, v1;
    if (more) {   // register-staged prefetch
      const uint4* g = (const uint4*)(kg + (size_t)(kb + 1) * 32 * CO);
      k0 = g[0]; k1 = g[1];
      const uint4* h = (const uint4*)(vg + (size_t)(kb + 1) * 32);
      v0 = h[0]; v1 = h[1];
    }
#endif

    // S^T tiles: rows = keys, cols = queries (per-lane query stats)
    v8f T[2];
#pragma unroll
    for (int kt = 0; kt < 2; ++kt) {
      v8f acc = {};
#pragma unroll
      for (int c2 = 0; c2 < 2; ++c2) {
        AF Af;
        int row = kt * 16 + l16;
        int eb  = c2 * 32 + (hi ? 8 : 0);
        Af.q[0] = *(const uint4*)&Kc[row * KS + eb];
        Af.q[1] = *(const uint4*)&Kc[row * KS + eb + 16];
        acc = __builtin_amdgcn_wmma_f32_16x16x32_f16(false, Af.h, false, Qb[c2].h,
                                                     (short)0, acc, false, false);
      }
      T[kt] = acc;
    }

    // online softmax
    float vmax = T[0][0];
#pragma unroll
    for (int j = 1; j < 8; ++j) vmax = fmaxf(vmax, T[0][j]);
#pragma unroll
    for (int j = 0; j < 8; ++j) vmax = fmaxf(vmax, T[1][j]);
    vmax = fmaxf(vmax, __shfl_xor(vmax, 16));
    float m_new = fmaxf(m_i, vmax);
    float scl = __expf(m_i - m_new);
    m_i = m_new;
    float rs = 0.f;
#pragma unroll
    for (int kt = 0; kt < 2; ++kt)
#pragma unroll
      for (int j = 0; j < 8; ++j) { float e = __expf(T[kt][j] - m_new); T[kt][j] = e; rs += e; }
    rs += __shfl_xor(rs, 16);
    l_i = l_i * scl + rs;
#pragma unroll
    for (int ot = 0; ot < 4; ++ot)
#pragma unroll
      for (int j = 0; j < 8; ++j) accO[ot][j] *= scl;

    // P^T (C/D layout) -> B-operand layout: exchange M-halves across lane halves
    AF Pb;
    float t0x[8], t1x[8];
#pragma unroll
    for (int j = 0; j < 8; ++j) { t0x[j] = __shfl_xor(T[0][j], 16); t1x[j] = __shfl_xor(T[1][j], 16); }
#pragma unroll
    for (int j = 0; j < 4; ++j)
      Pb.u[j] = hi ? pk2(t1x[2*j], t1x[2*j+1]) : pk2(T[0][2*j], T[0][2*j+1]);
#pragma unroll
    for (int j = 4; j < 8; ++j) {
      int jj = 2 * (j - 4);
      Pb.u[j] = hi ? pk2(T[1][jj], T[1][jj+1]) : pk2(t0x[jj], t0x[jj+1]);
    }

    // O^T += V^T * P^T
#pragma unroll
    for (int ot = 0; ot < 4; ++ot) {
      AF Af;
      int row = ot * 16 + l16;
      int eb  = hi ? 16 : 0;
      Af.q[0] = *(const uint4*)&Vc[row * VSS + eb];
      Af.q[1] = *(const uint4*)&Vc[row * VSS + eb + 8];
      accO[ot] = __builtin_amdgcn_wmma_f32_16x16x32_f16(false, Af.h, false, Pb.h,
                                                        (short)0, accO[ot], false, false);
    }

#if USE_ASYNC
    if (more) __builtin_amdgcn_s_wait_asynccnt(0);
#else
    if (more) {
      ((uint4*)kbufs[cur ^ 1])[0] = k0; ((uint4*)kbufs[cur ^ 1])[1] = k1;
      ((uint4*)vbufs[cur ^ 1])[0] = v0; ((uint4*)vbufs[cur ^ 1])[1] = v1;
    }
#endif
    __syncthreads();
  }

  // normalize, convert O^T to B-operand fragments
  float rinv = 1.0f / l_i;
  v8f On[4];
#pragma unroll
  for (int ot = 0; ot < 4; ++ot)
#pragma unroll
    for (int j = 0; j < 8; ++j) On[ot][j] = accO[ot][j] * rinv;
  AF Ob[2];
#pragma unroll
  for (int ch = 0; ch < 2; ++ch) {
    v8f& Ta = On[ch * 2]; v8f& Tb = On[ch * 2 + 1];
    float tax[8], tbx[8];
#pragma unroll
    for (int j = 0; j < 8; ++j) { tax[j] = __shfl_xor(Ta[j], 16); tbx[j] = __shfl_xor(Tb[j], 16); }
#pragma unroll
    for (int j = 0; j < 4; ++j)
      Ob[ch].u[j] = hi ? pk2(tbx[2*j], tbx[2*j+1]) : pk2(Ta[2*j], Ta[2*j+1]);
#pragma unroll
    for (int j = 4; j < 8; ++j) {
      int jj = 2 * (j - 4);
      Ob[ch].u[j] = hi ? pk2(Tb[jj], Tb[jj+1]) : pk2(tax[jj], tax[jj+1]);
    }
  }

  // y^T = wc * O^T ; BN scale/shift ; ReLU ; residual
  const int n = q0 + wave * 16 + l16;
#pragma unroll
  for (int ct = 0; ct < 8; ++ct) {
    v8f acc = {};
#pragma unroll
    for (int ch = 0; ch < 2; ++ch) {
      AF Af;
      int row = ct * 16 + l16;
      int eb  = ch * 32 + (hi ? 16 : 0);
      Af.q[0] = *(const uint4*)&wc_s[row * KS + eb];
      Af.q[1] = *(const uint4*)&wc_s[row * KS + eb + 8];
      acc = __builtin_amdgcn_wmma_f32_16x16x32_f16(false, Af.h, false, Ob[ch].h,
                                                   (short)0, acc, false, false);
    }
#pragma unroll
    for (int j = 0; j < 8; ++j) {
      int c = ct * 16 + j + (hi ? 8 : 0);
      float y = acc[j] * scale_s[c] + shift_s[c];
      y = fmaxf(y, 0.f);
      size_t idx = ((size_t)(b * CH + c)) * NN + n;
      out[idx] = img[idx] + y;
    }
  }
}

// ---------------------------------------------------------------------------
extern "C" void kernel_launch(void* const* d_in, const int* in_sizes, int n_in,
                              void* d_out, int out_size, void* d_ws, size_t ws_size,
                              hipStream_t stream) {
  (void)in_sizes; (void)n_in; (void)out_size; (void)ws_size;
  const float* range_x = (const float*)d_in[0];
  const float* img     = (const float*)d_in[1];
  const float* wq = (const float*)d_in[2];
  const float* bq = (const float*)d_in[3];
  const float* wk = (const float*)d_in[4];
  const float* bk = (const float*)d_in[5];
  const float* wv = (const float*)d_in[6];
  const float* bv = (const float*)d_in[7];
  const float* wc = (const float*)d_in[8];
  const float* bc = (const float*)d_in[9];
  const float* gamma = (const float*)d_in[10];
  const float* beta  = (const float*)d_in[11];
  const float* mean  = (const float*)d_in[12];
  const float* var   = (const float*)d_in[13];
  float* out = (float*)d_out;

  _Float16* Qh = (_Float16*)d_ws;                       // [B][N][64]
  _Float16* Kh = Qh + (size_t)NB * NN * CO;             // [B][N][64]
  _Float16* Vt = Kh + (size_t)NB * NN * CO;             // [B][64][N]

  proj_kernel<<<NB * (NN / 64), 128, 0, stream>>>(range_x, img, wq, bq, wk, bk, wv, bv,
                                                  Qh, Kh, Vt);
  flash_kernel<<<NB * (NN / 64), 128, 0, stream>>>(Qh, Kh, Vt, img, wc, bc,
                                                   gamma, beta, mean, var, out);
}
